// DG_89867895701825
// MI455X (gfx1250) — compile-verified, matrix operand order checked by reading, not run on hardware
//
#include <hip/hip_runtime.h>
#include <hip/hip_bf16.h>
#include <stdint.h>

// ---------------------------------------------------------------------------
// Problem constants (match reference)
// ---------------------------------------------------------------------------
#define BATCH      2048
#define INPUT_SIZE 4096
#define NUM_UNITS  2048
#define SPARSITY   20
#define DECAY      0.95f

typedef __attribute__((ext_vector_type(16))) __bf16 v16bf;
typedef __attribute__((ext_vector_type(8)))  float  v8f;

// ---------------------------------------------------------------------------
// Monotone float<->uint mapping (total order on floats as unsigned ints)
// ---------------------------------------------------------------------------
__device__ __forceinline__ uint32_t mono_f32(float f) {
    uint32_t b = __float_as_uint(f);
    return (b & 0x80000000u) ? ~b : (b | 0x80000000u);
}
__device__ __forceinline__ float demono_f32(uint32_t u) {
    uint32_t b = (u & 0x80000000u) ? (u & 0x7FFFFFFFu) : ~u;
    return __uint_as_float(b);
}

__device__ __forceinline__ v16bf ldfrag(const __bf16* p) {
    return *(const v16bf*)p;
}

// ---------------------------------------------------------------------------
// 1) init control slots (umax=0 encodes -inf side, umin=0xFFFFFFFF)
// ---------------------------------------------------------------------------
__global__ void DG_init_kernel(uint32_t* ctrl) {
    if (threadIdx.x == 0) {
        ctrl[0] = 0u;           // running max (monotone-encoded)
        ctrl[1] = 0xFFFFFFFFu;  // running min (monotone-encoded)
    }
}

// ---------------------------------------------------------------------------
// 2) global min/max of inputs (grid-stride + block reduce + atomics)
// ---------------------------------------------------------------------------
__global__ __launch_bounds__(256) void DG_minmax_kernel(const float* __restrict__ x,
                                                        long long n, uint32_t* ctrl) {
    __shared__ float smax[256];
    __shared__ float smin[256];
    int tid = threadIdx.x;
    long long i = (long long)blockIdx.x * blockDim.x + tid;
    long long stride = (long long)gridDim.x * blockDim.x;
    float vmax = -__builtin_inff();
    float vmin =  __builtin_inff();
    for (; i < n; i += stride) {
        float v = x[i];
        vmax = fmaxf(vmax, v);
        vmin = fminf(vmin, v);
    }
    smax[tid] = vmax; smin[tid] = vmin;
    __syncthreads();
    for (int off = 128; off > 0; off >>= 1) {
        if (tid < off) {
            smax[tid] = fmaxf(smax[tid], smax[tid + off]);
            smin[tid] = fminf(smin[tid], smin[tid + off]);
        }
        __syncthreads();
    }
    if (tid == 0) {
        atomicMax(&ctrl[0], mono_f32(smax[0]));
        atomicMin(&ctrl[1], mono_f32(smin[0]));
    }
}

// ---------------------------------------------------------------------------
// 3) finalize: decode min/max, store minv and scale s = 1/(max-min)
// ---------------------------------------------------------------------------
__global__ void DG_finalize_kernel(uint32_t* ctrl) {
    if (threadIdx.x == 0) {
        float maxv = demono_f32(ctrl[0]);
        float minv = demono_f32(ctrl[1]);
        float* f = (float*)ctrl;
        f[2] = minv;
        f[3] = 1.0f / (maxv - minv);
    }
}

// ---------------------------------------------------------------------------
// 4) f32 -> bf16 hi/lo split conversion (templated: no per-element branch)
// ---------------------------------------------------------------------------
template <int WRITE_LO>
__global__ __launch_bounds__(256) void DG_convert_kernel(const float* __restrict__ src,
                                                         __bf16* __restrict__ hi,
                                                         __bf16* __restrict__ lo,
                                                         int n) {
    int i = blockIdx.x * blockDim.x + threadIdx.x;
    int stride = gridDim.x * blockDim.x;
    for (; i < n; i += stride) {
        float v = src[i];
        __bf16 h = (__bf16)v;
        hi[i] = h;
        if (WRITE_LO) lo[i] = (__bf16)(v - (float)h);
    }
}

// ---------------------------------------------------------------------------
// 5) per-unit row sums of W (for the affine normalization correction)
// ---------------------------------------------------------------------------
__global__ __launch_bounds__(256) void DG_rowsum_kernel(const float* __restrict__ W,
                                                        float* __restrict__ rowsum) {
    __shared__ float s[256];
    int row = blockIdx.x, tid = threadIdx.x;
    const float* p = W + (size_t)row * INPUT_SIZE;
    float acc = 0.0f;
    for (int k = tid; k < INPUT_SIZE; k += 256) acc += p[k];
    s[tid] = acc;
    __syncthreads();
    for (int off = 128; off > 0; off >>= 1) {
        if (tid < off) s[tid] += s[tid + off];
        __syncthreads();
    }
    if (tid == 0) rowsum[row] = s[0];
}

// ---------------------------------------------------------------------------
// 6) WMMA GEMM: enc[b,u] = s * dot(in_b, W_u) - s*min*rowsum[u]
//    bf16 hi/lo split (hi*hi + hi*lo + lo*hi) for ~f32 accuracy.
//    Block = 256 threads = 8 waves arranged 2(M) x 4(N-pairs);
//    each wave owns TWO 16x16 N-tiles -> block tile 32x128,
//    6 fragment loads feed 6 WMMAs per K-step (1:1 ratio).
//    Explicit 2-stage register double-buffer: next-iteration fragment loads
//    are issued before the current WMMAs so loadcnt waits overlap matrix ops.
// ---------------------------------------------------------------------------
template <int USE_LO>
__global__ __launch_bounds__(256) void DG_gemm_kernel(const __bf16* __restrict__ Ah,
                                                      const __bf16* __restrict__ Al,
                                                      const __bf16* __restrict__ Bh,
                                                      const __bf16* __restrict__ Bl,
                                                      const float* __restrict__ ctrl,
                                                      const float* __restrict__ rowsumW,
                                                      float* __restrict__ enc) {
    const int K = INPUT_SIZE;
    int tid  = threadIdx.x;
    int lane = tid & 31;
    int w    = tid >> 5;

    int m0 = blockIdx.y * 32  + (w & 1)  * 16;   // batch tile
    int n0 = blockIdx.x * 128 + (w >> 1) * 32;   // unit tile pair: n0, n0+16

    int arow  = m0 + (lane & 15);
    int brow0 = n0 + (lane & 15);
    int brow1 = brow0 + 16;
    int koff  = (lane >> 4) * 16;  // lane group selects K half (same perm for A & B)

    const __bf16* aph  = Ah + (size_t)arow  * K + koff;
    const __bf16* apl  = Al + (size_t)arow  * K + koff;
    const __bf16* bph0 = Bh + (size_t)brow0 * K + koff;
    const __bf16* bpl0 = Bl + (size_t)brow0 * K + koff;
    const __bf16* bph1 = Bh + (size_t)brow1 * K + koff;
    const __bf16* bpl1 = Bl + (size_t)brow1 * K + koff;

    v8f acc0 = {}, acc1 = {};

    // stage-0 preload
    v16bf a  = ldfrag(aph);
    v16bf b0 = ldfrag(bph0);
    v16bf b1 = ldfrag(bph1);
    v16bf al = {}, bl0 = {}, bl1 = {};
    if (USE_LO) { al = ldfrag(apl); bl0 = ldfrag(bpl0); bl1 = ldfrag(bpl1); }

    for (int k = 32; k < K; k += 32) {
        // issue next-iteration loads first (overlap with WMMAs below)
        v16bf an  = ldfrag(aph  + k);
        v16bf b0n = ldfrag(bph0 + k);
        v16bf b1n = ldfrag(bph1 + k);
        v16bf aln = {}, bl0n = {}, bl1n = {};
        if (USE_LO) { aln = ldfrag(apl + k); bl0n = ldfrag(bpl0 + k); bl1n = ldfrag(bpl1 + k); }
        if (k + 32 < K) {
            __builtin_prefetch(aph  + k + 32, 0, 1);
            __builtin_prefetch(bph0 + k + 32, 0, 1);
            __builtin_prefetch(bph1 + k + 32, 0, 1);
        }

        acc0 = __builtin_amdgcn_wmma_f32_16x16x32_bf16(false, a, false, b0,
                                                       (short)0, acc0, false, false);
        acc1 = __builtin_amdgcn_wmma_f32_16x16x32_bf16(false, a, false, b1,
                                                       (short)0, acc1, false, false);
        if (USE_LO) {
            acc0 = __builtin_amdgcn_wmma_f32_16x16x32_bf16(false, a,  false, bl0,
                                                           (short)0, acc0, false, false);
            acc0 = __builtin_amdgcn_wmma_f32_16x16x32_bf16(false, al, false, b0,
                                                           (short)0, acc0, false, false);
            acc1 = __builtin_amdgcn_wmma_f32_16x16x32_bf16(false, a,  false, bl1,
                                                           (short)0, acc1, false, false);
            acc1 = __builtin_amdgcn_wmma_f32_16x16x32_bf16(false, al, false, b1,
                                                           (short)0, acc1, false, false);
        }

        a = an; b0 = b0n; b1 = b1n;
        if (USE_LO) { al = aln; bl0 = bl0n; bl1 = bl1n; }
    }

    // epilogue iteration (fragments already resident)
    acc0 = __builtin_amdgcn_wmma_f32_16x16x32_bf16(false, a, false, b0,
                                                   (short)0, acc0, false, false);
    acc1 = __builtin_amdgcn_wmma_f32_16x16x32_bf16(false, a, false, b1,
                                                   (short)0, acc1, false, false);
    if (USE_LO) {
        acc0 = __builtin_amdgcn_wmma_f32_16x16x32_bf16(false, a,  false, bl0,
                                                       (short)0, acc0, false, false);
        acc0 = __builtin_amdgcn_wmma_f32_16x16x32_bf16(false, al, false, b0,
                                                       (short)0, acc0, false, false);
        acc1 = __builtin_amdgcn_wmma_f32_16x16x32_bf16(false, a,  false, bl1,
                                                       (short)0, acc1, false, false);
        acc1 = __builtin_amdgcn_wmma_f32_16x16x32_bf16(false, al, false, b1,
                                                       (short)0, acc1, false, false);
    }

    // fused normalization epilogue: enc = s*dot + beta[unit]
    float minv = ctrl[2];
    float s    = ctrl[3];
    int nA = n0 + (lane & 15);
    int nB = nA + 16;
    float betaA = -s * minv * rowsumW[nA];
    float betaB = -s * minv * rowsumW[nB];
    int mbase = m0 + ((lane >> 4) << 3);  // lanes 16-31 hold M = r+8
    for (int r = 0; r < 8; ++r) {
        size_t rowoff = (size_t)(mbase + r) * NUM_UNITS;
        enc[rowoff + nA] = s * acc0[r] + betaA;
        enc[rowoff + nB] = s * acc1[r] + betaB;
    }
}

// ---------------------------------------------------------------------------
// 7) Exact top-k mask over 2048 values (256 threads x 8 values/thread).
//    4-pass 8-bit radix select on monotone keys + index-ordered tie fill.
//    Matches jax.lax.top_k semantics (ties -> lowest index).
// ---------------------------------------------------------------------------
__device__ void topk_mask_256x8(const uint32_t keys[8], int k,
                                uint32_t* hist, uint32_t* scanbuf,
                                volatile uint32_t* scal,
                                int tid, bool mask_out[8]) {
    uint32_t prefmask = 0u, prefval = 0u;
    uint32_t remaining = (uint32_t)k;
    uint32_t ceq_last = 0u;

    for (int p = 3; p >= 0; --p) {
        hist[tid] = 0u;
        __syncthreads();
        for (int j = 0; j < 8; ++j) {
            if ((keys[j] & prefmask) == prefval) {
                atomicAdd(&hist[(keys[j] >> (p * 8)) & 0xFFu], 1u);
            }
        }
        __syncthreads();
        // inclusive suffix sum of hist into scanbuf
        scanbuf[tid] = hist[tid];
        __syncthreads();
        for (int off = 1; off < 256; off <<= 1) {
            uint32_t add = (tid + off < 256) ? scanbuf[tid + off] : 0u;
            __syncthreads();
            scanbuf[tid] += add;
            __syncthreads();
        }
        // pick crossing bucket
        uint32_t sfx  = scanbuf[tid];
        uint32_t snxt = (tid < 255) ? scanbuf[tid + 1] : 0u;
        if (sfx >= remaining && snxt < remaining) {
            scal[0] = (uint32_t)tid;          // selected bucket
            scal[1] = remaining - snxt;       // rank left inside bucket
            scal[2] = hist[tid];              // candidates equal at this byte
        }
        __syncthreads();
        uint32_t bucket = scal[0];
        remaining       = scal[1];
        ceq_last        = scal[2];
        prefmask |= 0xFFu << (p * 8);
        prefval  |= bucket << (p * 8);
        __syncthreads();
    }

    uint32_t T = prefval;  // exact k-th largest key
    if (ceq_last == remaining) {
        // all keys equal to threshold fire: no tie-break needed
        for (int j = 0; j < 8; ++j) mask_out[j] = (keys[j] >= T);
    } else {
        // index-ordered tie fill: exclusive prefix sum of per-thread equal counts
        uint32_t eqc = 0u;
        for (int j = 0; j < 8; ++j) eqc += (keys[j] == T) ? 1u : 0u;
        scanbuf[tid] = eqc;
        __syncthreads();
        for (int off = 1; off < 256; off <<= 1) {
            uint32_t add = (tid >= off) ? scanbuf[tid - off] : 0u;
            __syncthreads();
            scanbuf[tid] += add;
            __syncthreads();
        }
        uint32_t taken = scanbuf[tid] - eqc;  // equals with smaller global index
        for (int j = 0; j < 8; ++j) {
            if (keys[j] == T) {
                mask_out[j] = (taken < remaining);
                taken++;
            } else {
                mask_out[j] = (keys[j] > T);
            }
        }
        __syncthreads();
    }
}

// ---------------------------------------------------------------------------
// 8) Sequential lateral-inhibition scan over batch rows (single workgroup).
// ---------------------------------------------------------------------------
__global__ __launch_bounds__(256) void DG_scan_kernel(const float* __restrict__ enc,
                                                      float* __restrict__ out) {
    __shared__ uint32_t hist[256];
    __shared__ uint32_t scanbuf[256];
    __shared__ uint32_t scal[4];
    __shared__ int flagNF;

    int tid = threadIdx.x;
    float inhib[8];
    for (int j = 0; j < 8; ++j) inhib[j] = 0.0f;

    for (int r = 0; r < BATCH; ++r) {
        const float* row = enc + (size_t)r * NUM_UNITS + tid * 8;
        float e[8];
        uint32_t keys[8];
        for (int j = 0; j < 8; ++j) e[j] = row[j];
        for (int j = 0; j < 8; ++j) {
            float refr = fabsf(e[j]) * (1.0f - inhib[j]);
            keys[j] = mono_f32(refr);
        }

        bool m[8];
        topk_mask_256x8(keys, SPARSITY, hist, scanbuf, scal, tid, m);

        // fast path: if every fired value is strictly positive, the final
        // top-k over (enc * mask) equals the step mask exactly.
        if (tid == 0) flagNF = 0;
        __syncthreads();
        int bad = 0;
        for (int j = 0; j < 8; ++j)
            if (m[j] && !(e[j] > 0.0f)) bad = 1;
        if (bad) atomicOr(&flagNF, 1);
        __syncthreads();

        bool fm[8];
        if (flagNF) {
            uint32_t k2[8];
            for (int j = 0; j < 8; ++j) k2[j] = mono_f32(m[j] ? e[j] : 0.0f);
            __syncthreads();
            topk_mask_256x8(k2, SPARSITY, hist, scanbuf, scal, tid, fm);
        } else {
            for (int j = 0; j < 8; ++j) fm[j] = m[j];
        }

        float* orow = out + (size_t)r * NUM_UNITS + tid * 8;
        for (int j = 0; j < 8; ++j) orow[j] = fm[j] ? 1.0f : 0.0f;
        for (int j = 0; j < 8; ++j)
            inhib[j] = inhib[j] * DECAY + (m[j] ? 1.0f : 0.0f);
        __syncthreads();
    }
}

// ---------------------------------------------------------------------------
// Workspace layout (bytes)
// ---------------------------------------------------------------------------
#define WS_CTRL    0                                     // 4 x u32/f32
#define WS_ROWSUM  1024                                  // 2048 f32 = 8 KB
#define WS_AHI     65536                                 // 8M bf16 = 16 MB
#define WS_BHI     (WS_AHI + (size_t)BATCH * INPUT_SIZE * 2)
#define WS_ENC     (WS_BHI + (size_t)NUM_UNITS * INPUT_SIZE * 2)
#define WS_ALO     (WS_ENC + (size_t)BATCH * NUM_UNITS * 4)
#define WS_BLO     (WS_ALO + (size_t)BATCH * INPUT_SIZE * 2)
#define WS_END     (WS_BLO + (size_t)NUM_UNITS * INPUT_SIZE * 2)

extern "C" void kernel_launch(void* const* d_in, const int* in_sizes, int n_in,
                              void* d_out, int out_size, void* d_ws, size_t ws_size,
                              hipStream_t stream) {
    (void)in_sizes; (void)n_in; (void)out_size;
    const float* inputs = (const float*)d_in[0];  // [2048, 4096]
    const float* W      = (const float*)d_in[1];  // [2048, 4096]
    float* out          = (float*)d_out;          // [2048, 2048]

    char* ws = (char*)d_ws;
    uint32_t* ctrl  = (uint32_t*)(ws + WS_CTRL);
    float* rowsum   = (float*)(ws + WS_ROWSUM);
    __bf16* Ah      = (__bf16*)(ws + WS_AHI);
    __bf16* Bh      = (__bf16*)(ws + WS_BHI);
    float* enc      = (float*)(ws + WS_ENC);
    __bf16* Al      = (__bf16*)(ws + WS_ALO);
    __bf16* Bl      = (__bf16*)(ws + WS_BLO);

    int use_lo = (ws_size >= WS_END) ? 1 : 0;
    if (!use_lo) { Al = Ah; Bl = Bh; }  // unused but keep pointers valid

    const long long nIn = (long long)BATCH * INPUT_SIZE;
    const int nW = NUM_UNITS * INPUT_SIZE;

    // 1. init control slots
    DG_init_kernel<<<1, 32, 0, stream>>>(ctrl);
    // 2. global min/max of inputs
    DG_minmax_kernel<<<1024, 256, 0, stream>>>(inputs, nIn, ctrl);
    // 3. finalize min/scale
    DG_finalize_kernel<<<1, 32, 0, stream>>>(ctrl);
    // 4. bf16 hi/lo splits
    if (use_lo) {
        DG_convert_kernel<1><<<4096, 256, 0, stream>>>(inputs, Ah, Al, (int)nIn);
        DG_convert_kernel<1><<<4096, 256, 0, stream>>>(W, Bh, Bl, nW);
    } else {
        DG_convert_kernel<0><<<4096, 256, 0, stream>>>(inputs, Ah, Al, (int)nIn);
        DG_convert_kernel<0><<<4096, 256, 0, stream>>>(W, Bh, Bl, nW);
    }
    // 5. W row sums for the affine correction
    DG_rowsum_kernel<<<NUM_UNITS, 256, 0, stream>>>(W, rowsum);
    // 6. WMMA GEMM with fused normalization epilogue (block tile 32x128)
    dim3 ggrid(NUM_UNITS / 128, BATCH / 32);
    if (use_lo) {
        DG_gemm_kernel<1><<<ggrid, 256, 0, stream>>>(Ah, Al, Bh, Bl,
                                                     (const float*)ctrl, rowsum, enc);
    } else {
        DG_gemm_kernel<0><<<ggrid, 256, 0, stream>>>(Ah, Al, Bh, Bl,
                                                     (const float*)ctrl, rowsum, enc);
    }
    // 7. sequential inhibition scan + exact top-k masks
    DG_scan_kernel<<<1, 256, 0, stream>>>(enc, out);
}
